// Mamba_19868518711659
// MI455X (gfx1250) — matrix operand order repaired and checked by reading, not compiled, not run on hardware
//
#include <hip/hip_runtime.h>
#include <math.h>

typedef __attribute__((ext_vector_type(16))) _Float16 v16h;
typedef __attribute__((ext_vector_type(8)))  float    v8f;

namespace {
constexpr int B_ = 2, S_ = 256, CN_ = 128, P_ = 16, D_ = 512, DI_ = 1024,
              DS_ = 16, DC_ = 4, DT_ = 32, NB_ = 4;
constexpr int L_  = (S_ - P_) / P_ + 1;   // 16 patches
constexpr int M_  = B_ * CN_ * L_;        // 4096 token rows (b,c,l)
constexpr int BC_ = B_ * CN_;             // 256
constexpr int NX_ = DT_ + 2 * DS_;        // 64 (xproj out)
constexpr int KP_ = 32;                   // embed K padded 16 -> 32
constexpr float EPS_ = 1e-5f;
}

__device__ __forceinline__ float siluf(float x) { return x / (1.f + __expf(-x)); }
__device__ __forceinline__ float softplusf(float x) {
  return (x > 20.f) ? x : log1pf(__expf(x));
}

// A-fragment: halves j=0..15 <- p[0..15] (16 contiguous K values)
__device__ __forceinline__ v16h loadA_frag(const float* p) {
  const float4 x0 = *reinterpret_cast<const float4*>(p);
  const float4 x1 = *reinterpret_cast<const float4*>(p + 4);
  const float4 x2 = *reinterpret_cast<const float4*>(p + 8);
  const float4 x3 = *reinterpret_cast<const float4*>(p + 12);
  v16h a;
  a[0]  = (_Float16)x0.x; a[1]  = (_Float16)x0.y;
  a[2]  = (_Float16)x0.z; a[3]  = (_Float16)x0.w;
  a[4]  = (_Float16)x1.x; a[5]  = (_Float16)x1.y;
  a[6]  = (_Float16)x1.z; a[7]  = (_Float16)x1.w;
  a[8]  = (_Float16)x2.x; a[9]  = (_Float16)x2.y;
  a[10] = (_Float16)x2.z; a[11] = (_Float16)x2.w;
  a[12] = (_Float16)x3.x; a[13] = (_Float16)x3.y;
  a[14] = (_Float16)x3.z; a[15] = (_Float16)x3.w;
  return a;
}

// B-fragment: halves j=0..15 <- p[0..15] (16 contiguous K values)
__device__ __forceinline__ v16h loadB_frag(const float* p) {
  const float4 x0 = *reinterpret_cast<const float4*>(p);
  const float4 x1 = *reinterpret_cast<const float4*>(p + 4);
  const float4 x2 = *reinterpret_cast<const float4*>(p + 8);
  const float4 x3 = *reinterpret_cast<const float4*>(p + 12);
  v16h b;
  b[0]  = (_Float16)x0.x; b[1]  = (_Float16)x0.y;
  b[2]  = (_Float16)x0.z; b[3]  = (_Float16)x0.w;
  b[4]  = (_Float16)x1.x; b[5]  = (_Float16)x1.y;
  b[6]  = (_Float16)x1.z; b[7]  = (_Float16)x1.w;
  b[8]  = (_Float16)x2.x; b[9]  = (_Float16)x2.y;
  b[10] = (_Float16)x2.z; b[11] = (_Float16)x2.w;
  b[12] = (_Float16)x3.x; b[13] = (_Float16)x3.y;
  b[14] = (_Float16)x3.z; b[15] = (_Float16)x3.w;
  return b;
}

// ---------------------------------------------------------------------------
// GEMM: C[M,N] = A[M,K] * Bw[N,K]^T + bias[N]
// Requirements (all satisfied in this model): K%32==0, N%64==0, M%32==0.
// fp32 in memory, f16 fragments, fp32 WMMA accumulation.
// Block = 256 threads = 8 waves; wave tile = 16(M) x 64(N); block = 32 x 256.
// ---------------------------------------------------------------------------
__global__ __launch_bounds__(256) void wmma_gemm_bias(
    const float* __restrict__ A, int lda,
    const float* __restrict__ Bw, int ldb,
    const float* __restrict__ bias,
    float* __restrict__ C, int ldc,
    int M, int N, int K)
{
  const int tid   = threadIdx.x;
  const int lane  = tid & 31;
  const int wave  = __builtin_amdgcn_readfirstlane(tid >> 5);   // wave-uniform SGPR
  const int baseM = blockIdx.y * 32 + (wave & 1) * 16;
  const int baseN = blockIdx.x * 256 + (wave >> 1) * 64;
  if (baseM >= M || baseN >= N) return;     // scalar branch, EXEC untouched

  const int  row = lane & 15;
  const bool hi  = lane >= 16;

  // Streaming pointers (aligned: lda/ldb multiples of 8 floats, offsets of 8/16)
  const float* ap  = A  + (size_t)(baseM + row) * lda + (hi ? 16 : 0);
  const float* bp0 = Bw + (size_t)(baseN + row) * ldb + (hi ? 16 : 0);
  const float* bp1 = bp0 + (size_t)16 * ldb;
  const float* bp2 = bp0 + (size_t)32 * ldb;
  const float* bp3 = bp0 + (size_t)48 * ldb;

  v8f acc0 = {}, acc1 = {}, acc2 = {}, acc3 = {};
#pragma unroll 2
  for (int kt = 0; kt < K; kt += 32) {
    const v16h a  = loadA_frag(ap);
    const v16h b0 = loadB_frag(bp0);
    const v16h b1 = loadB_frag(bp1);
    const v16h b2 = loadB_frag(bp2);
    const v16h b3 = loadB_frag(bp3);
    acc0 = __builtin_amdgcn_wmma_f32_16x16x32_f16(false, a, false, b0, (short)0, acc0, false, false);
    acc1 = __builtin_amdgcn_wmma_f32_16x16x32_f16(false, a, false, b1, (short)0, acc1, false, false);
    acc2 = __builtin_amdgcn_wmma_f32_16x16x32_f16(false, a, false, b2, (short)0, acc2, false, false);
    acc3 = __builtin_amdgcn_wmma_f32_16x16x32_f16(false, a, false, b3, (short)0, acc3, false, false);
    ap += 32; bp0 += 32; bp1 += 32; bp2 += 32; bp3 += 32;
  }

  // D layout: VGPR r -> row r (lanes 0-15) / r+8 (lanes 16-31); col = lane&15.
  const int m0 = baseM + (hi ? 8 : 0);
  const int n0 = baseN + row;
  const float bv0 = bias ? bias[n0]      : 0.f;
  const float bv1 = bias ? bias[n0 + 16] : 0.f;
  const float bv2 = bias ? bias[n0 + 32] : 0.f;
  const float bv3 = bias ? bias[n0 + 48] : 0.f;
#pragma unroll
  for (int r = 0; r < 8; ++r) {
    float* cr = C + (size_t)(m0 + r) * ldc;
    cr[n0]      = acc0[r] + bv0;
    cr[n0 + 16] = acc1[r] + bv1;
    cr[n0 + 32] = acc2[r] + bv2;
    cr[n0 + 48] = acc3[r] + bv3;
  }
}

// pat[(b,c,l), p] = x[b, l*P+p, c] for p<16; zero pad p=16..31 (K padded to 32)
__global__ void patchify_pad_kernel(const float* __restrict__ x,
                                    float* __restrict__ pat) {
  int idx = blockIdx.x * blockDim.x + threadIdx.x;
  if (idx >= M_ * KP_) return;
  int p = idx & (KP_ - 1);
  int m = idx >> 5;
  int l = m % L_;
  int c = (m / L_) % CN_;
  int b = m / (L_ * CN_);
  pat[idx] = (p < P_) ? x[((size_t)(b * S_ + l * P_ + p)) * CN_ + c] : 0.f;
}

// emb_w [D,16] -> zero-padded [D,32]
__global__ void pad_embw_kernel(const float* __restrict__ w,
                                float* __restrict__ wp) {
  int idx = blockIdx.x * blockDim.x + threadIdx.x;
  if (idx >= D_ * KP_) return;
  int p = idx & (KP_ - 1);
  int d = idx >> 5;
  wp[idx] = (p < P_) ? w[d * P_ + p] : 0.f;
}

// causal depthwise conv over l (kernel DC) + SiLU; xs = xr[:, 0:DI]
__global__ void conv_silu_kernel(const float* __restrict__ xr,
                                 const float* __restrict__ cw,
                                 const float* __restrict__ cb,
                                 float* __restrict__ xc) {
  int idx = blockIdx.x * blockDim.x + threadIdx.x;
  if (idx >= M_ * DI_) return;
  int di = idx & (DI_ - 1);
  int m  = idx >> 10;          // DI_ == 1024
  int l  = m % L_;
  float acc = cb[di];
#pragma unroll
  for (int k = 0; k < DC_; ++k) {
    int ls = l + k - (DC_ - 1);
    if (ls >= 0)
      acc += cw[di * DC_ + k] * xr[(size_t)(m + k - (DC_ - 1)) * (2 * DI_) + di];
  }
  xc[idx] = siluf(acc);
}

// selective scan over L steps; state [DS] per (b,c,di) in registers.
// softplus(dt), SSM recurrence, + skip Dp*u, gated by silu(res).
__global__ void scan_kernel(const float* __restrict__ dtb,
                            const float* __restrict__ xc,
                            const float* __restrict__ dbc,
                            const float* __restrict__ xr,
                            const float* __restrict__ A_log,
                            const float* __restrict__ Dp,
                            float* __restrict__ y) {
  int idx = blockIdx.x * blockDim.x + threadIdx.x;
  if (idx >= BC_ * DI_) return;
  int di = idx & (DI_ - 1);
  int bc = idx >> 10;
  float a[DS_], st[DS_];
#pragma unroll
  for (int s = 0; s < DS_; ++s) {
    a[s]  = -__expf(A_log[di * DS_ + s]);
    st[s] = 0.f;
  }
  const float Dpv = Dp[di];
  for (int l = 0; l < L_; ++l) {
    size_t m  = (size_t)bc * L_ + l;
    float  dt = softplusf(dtb[m * DI_ + di]);
    float  u  = xc[m * DI_ + di];
    float  du = dt * u;
    const float* Bp = dbc + m * NX_ + DT_;
    const float* Cp = Bp + DS_;
    float ys = 0.f;
#pragma unroll
    for (int s = 0; s < DS_; ++s) {
      st[s] = __expf(dt * a[s]) * st[s] + du * Bp[s];
      ys += st[s] * Cp[s];
    }
    float res = xr[m * (2 * DI_) + DI_ + di];
    y[m * DI_ + di] = (ys + Dpv * u) * siluf(res);
  }
}

// h += upd; LayerNorm jointly over (L,D) per (b,c); elementwise ln_w/ln_b.
__global__ __launch_bounds__(256) void residual_ln_kernel(
    float* __restrict__ h, const float* __restrict__ upd,
    const float* __restrict__ lnw, const float* __restrict__ lnb) {
  __shared__ float buf[L_ * D_];      // 32 KB of the 320 KB WGP LDS
  __shared__ float rs[256], rs2[256];
  const int    tid  = threadIdx.x;
  const size_t base = (size_t)blockIdx.x * (L_ * D_);
  float s = 0.f, s2 = 0.f;
  for (int j = tid; j < L_ * D_; j += 256) {
    float v = h[base + j] + upd[base + j];
    buf[j] = v; s += v; s2 += v * v;
  }
  rs[tid] = s; rs2[tid] = s2;
  __syncthreads();
  for (int off = 128; off > 0; off >>= 1) {
    if (tid < off) { rs[tid] += rs[tid + off]; rs2[tid] += rs2[tid + off]; }
    __syncthreads();
  }
  const float mean = rs[0] * (1.f / (L_ * D_));
  const float var  = rs2[0] * (1.f / (L_ * D_)) - mean * mean;
  const float inv  = rsqrtf(var + EPS_);
  for (int j = tid; j < L_ * D_; j += 256)
    h[base + j] = (buf[j] - mean) * inv * lnw[j] + lnb[j];
}

// d_out[b,s,c] = preds[(b,c), s]
__global__ void out_transpose_kernel(const float* __restrict__ preds,
                                     float* __restrict__ out) {
  int idx = blockIdx.x * blockDim.x + threadIdx.x;
  if (idx >= B_ * S_ * CN_) return;
  int c = idx % CN_;
  int s = (idx / CN_) % S_;
  int b = idx / (CN_ * S_);
  out[idx] = preds[((size_t)(b * CN_ + c)) * S_ + s];
}

extern "C" void kernel_launch(void* const* d_in, const int* in_sizes, int n_in,
                              void* d_out, int out_size, void* d_ws, size_t ws_size,
                              hipStream_t stream) {
  (void)in_sizes; (void)n_in; (void)out_size; (void)ws_size;
  const float* x      = (const float*)d_in[0];
  const float* emb_w  = (const float*)d_in[1];
  const float* emb_b  = (const float*)d_in[2];
  const float* in_w   = (const float*)d_in[3];
  const float* in_b   = (const float*)d_in[4];
  const float* conv_w = (const float*)d_in[5];
  const float* conv_b = (const float*)d_in[6];
  const float* xproj_w= (const float*)d_in[7];
  const float* dt_w   = (const float*)d_in[8];
  const float* dt_b   = (const float*)d_in[9];
  const float* A_log  = (const float*)d_in[10];
  const float* Dp     = (const float*)d_in[11];
  const float* out_w  = (const float*)d_in[12];
  const float* out_b  = (const float*)d_in[13];
  const float* ln_w   = (const float*)d_in[14];
  const float* ln_b   = (const float*)d_in[15];
  const float* step_w = (const float*)d_in[16];
  const float* step_b = (const float*)d_in[17];

  char*  ws  = (char*)d_ws;
  size_t off = 0;
  auto alloc = [&](size_t n) {
    float* p = (float*)(ws + off);
    off += n * sizeof(float);
    return p;
  };
  float* h     = alloc((size_t)M_ * D_);        // [4096,512]
  float* xr    = alloc((size_t)M_ * 2 * DI_);   // [4096,2048]
  float* xcb   = alloc((size_t)M_ * DI_);       // [4096,1024]
  float* dbc   = alloc((size_t)M_ * NX_);       // [4096,64]
  float* dtb   = alloc((size_t)M_ * DI_);       // [4096,1024]
  float* yb    = alloc((size_t)M_ * DI_);       // [4096,1024]
  float* upd   = alloc((size_t)M_ * D_);        // [4096,512]
  float* patP  = alloc((size_t)M_ * KP_);       // [4096,32] zero-padded
  float* embP  = alloc((size_t)D_ * KP_);       // [512,32]  zero-padded
  float* preds = alloc((size_t)BC_ * S_);       // [256,256]

  auto gemm = [&](const float* A, int lda, const float* Bw, int ldb,
                  const float* bias, float* C, int ldc, int M, int N, int K) {
    dim3 grid((unsigned)((N + 255) / 256), (unsigned)((M + 31) / 32));
    wmma_gemm_bias<<<grid, dim3(256), 0, stream>>>(A, lda, Bw, ldb, bias, C,
                                                   ldc, M, N, K);
  };

  // ---- patch embedding (K padded 16 -> 32) ----
  patchify_pad_kernel<<<(M_ * KP_ + 255) / 256, 256, 0, stream>>>(x, patP);
  pad_embw_kernel<<<(D_ * KP_ + 255) / 256, 256, 0, stream>>>(emb_w, embP);
  gemm(patP, KP_, embP, KP_, emb_b, h, D_, M_, D_, KP_);

  // ---- Mamba blocks ----
  for (int i = 0; i < NB_; ++i) {
    const float* in_w_i    = in_w    + (size_t)i * 2 * DI_ * D_;
    const float* in_b_i    = in_b    + (size_t)i * 2 * DI_;
    const float* conv_w_i  = conv_w  + (size_t)i * DI_ * DC_;
    const float* conv_b_i  = conv_b  + (size_t)i * DI_;
    const float* xproj_w_i = xproj_w + (size_t)i * NX_ * DI_;
    const float* dt_w_i    = dt_w    + (size_t)i * DI_ * DT_;
    const float* dt_b_i    = dt_b    + (size_t)i * DI_;
    const float* A_log_i   = A_log   + (size_t)i * DI_ * DS_;
    const float* Dp_i      = Dp      + (size_t)i * DI_;
    const float* out_w_i   = out_w   + (size_t)i * D_ * DI_;
    const float* out_b_i   = out_b   + (size_t)i * D_;
    const float* ln_w_i    = ln_w    + (size_t)i * L_ * D_;
    const float* ln_b_i    = ln_b    + (size_t)i * L_ * D_;

    // in_proj: [4096,512] x [2048,512]^T -> xr
    gemm(h, D_, in_w_i, D_, in_b_i, xr, 2 * DI_, M_, 2 * DI_, D_);
    // depthwise causal conv + silu -> xc
    conv_silu_kernel<<<(M_ * DI_ + 255) / 256, 256, 0, stream>>>(
        xr, conv_w_i, conv_b_i, xcb);
    // x_proj: [4096,1024] x [64,1024]^T -> dbc (dt_in | B | C)
    gemm(xcb, DI_, xproj_w_i, DI_, nullptr, dbc, NX_, M_, NX_, DI_);
    // dt_proj: dbc[:, :32] x [1024,32]^T + dt_b -> dtb (softplus in scan)
    gemm(dbc, NX_, dt_w_i, DT_, dt_b_i, dtb, DI_, M_, DI_, DT_);
    // selective scan + skip + gate -> yb
    scan_kernel<<<(BC_ * DI_ + 255) / 256, 256, 0, stream>>>(
        dtb, xcb, dbc, xr, A_log_i, Dp_i, yb);
    // out_proj: [4096,1024] x [512,1024]^T -> upd
    gemm(yb, DI_, out_w_i, DI_, out_b_i, upd, D_, M_, D_, DI_);
    // residual + joint LayerNorm over (L,D) per (b,c)
    residual_ln_kernel<<<BC_, 256, 0, stream>>>(h, upd, ln_w_i, ln_b_i);
  }

  // ---- final head: h viewed as [256, 8192] x step_w[256,8192]^T ----
  gemm(h, L_ * D_, step_w, L_ * D_, step_b, preds, S_, BC_, S_, L_ * D_);
  out_transpose_kernel<<<(B_ * S_ * CN_ + 255) / 256, 256, 0, stream>>>(
      preds, (float*)d_out);
}
